// GraphSensorFusion_76055280877926
// MI455X (gfx1250) — matrix operand order repaired and checked by reading, not compiled, not run on hardware
//
#include <hip/hip_runtime.h>
#include <cstdint>

typedef unsigned int uint32;
typedef uint32 u32x8 __attribute__((ext_vector_type(8)));
typedef __bf16  v16bf __attribute__((ext_vector_type(16)));
typedef float   v8f   __attribute__((ext_vector_type(8)));

#define BGRAPH 262144
#define SUBT 2            // graph-tiles per wave (B-frag reuse)
// per-sub-tile LDS staging region
#define CM_OFF 0          // col-major bf16 xw (128 cols * 48B)
#define RM_OFF 6144       // row-major bf16 x2/x3 (16 rows * 272B)
#define SC_OFF 10496      // score staging f32 (16 cols * 68B)
#define ZP_OFF 11584      // 32B of zeros (only region s=0 of each wave)
#define WAVE_LDS 11648
#define RM_PITCH 272
#define CM_STRIDE 48
#define SC_STRIDE 68

// RNE pack (prologue only: weights converted once, keep best rounding)
__device__ __forceinline__ uint32 pkbf_rne(float a, float b){
  uint32 ua = __float_as_uint(a), ub = __float_as_uint(b);
  ua = (ua + 0x7FFFu + ((ua>>16)&1u)) >> 16;
  ub = (ub + 0x7FFFu + ((ub>>16)&1u)) >> 16;
  return ua | (ub<<16);
}
// RTZ pack: single v_perm_b32 (result = lo(a>>16) | hi(b&0xffff0000))
__device__ __forceinline__ uint32 pkbf(float a, float b){
  return __builtin_amdgcn_perm(__float_as_uint(b), __float_as_uint(a), 0x07060302u);
}
__device__ __forceinline__ unsigned short bf16b(float a){
  return (unsigned short)(__float_as_uint(a) >> 16);   // RTZ, 1 VALU
}
__device__ __forceinline__ u32x8 mk8(uint4 a, uint4 b){
  u32x8 r; r[0]=a.x;r[1]=a.y;r[2]=a.z;r[3]=a.w;r[4]=b.x;r[5]=b.y;r[6]=b.z;r[7]=b.w;
  return r;
}
__device__ __forceinline__ v8f wmma_bf16(u32x8 a, u32x8 b, v8f c){
  return __builtin_amdgcn_wmma_f32_16x16x32_bf16(false, __builtin_bit_cast(v16bf,a),
                                                 false, __builtin_bit_cast(v16bf,b),
                                                 (short)0, c, false, false);
}
__device__ __forceinline__ void fence_lds(){
  __builtin_amdgcn_wave_barrier();
  asm volatile("" ::: "memory");
}

// ---------------- prologue: score vectors + B-fragment-linear bf16 weight images ----------------
// frag f (1KB): lane l holds 8 dwords at ws[f*256 + l*8 + d]
// B-frag layout: dword v: lanes0-15 -> K=2v+(v>>2)*8,+1 col=l ; lanes16-31 -> K=8+2v+(v>>2)*8,+1 col=l-16
// f 0..17  = B1ext [64 x 144] (cols 0..127 = W1^T, 128..131 = h0src,h0dst,h1src,h1dst), t*2+kf
// f 18..37 = B2ext [128 x 80] (cols 0..63 = W2^T, 64=w2src, 65=w2dst),                 18+t*4+kf
// f 38..53 = Bp    [64 x 128],                                                         38+t*2+kf
__global__ __launch_bounds__(512) void gat_prep_kernel(
    const float* __restrict__ W1, const float* __restrict__ a1s, const float* __restrict__ a1d,
    const float* __restrict__ W2, const float* __restrict__ a2s, const float* __restrict__ a2d,
    const float* __restrict__ Wp, uint32* __restrict__ ws)
{
  __shared__ float sv[512]; // 0:w1s0 64:w1d0 128:w1s1 192:w1d1 256:w2s(128) 384:w2d(128)
  int tid = threadIdx.x;
  if (tid < 256){
    int v = tid>>6, k = tid&63;
    const float* av = (v&1) ? a1d : a1s; av += (v>>1)*64;
    const float* wrow = W1 + ((v>>1)*64)*64 + k;
    float acc = 0.f;
    for(int o=0;o<64;++o) acc += av[o]*wrow[o*64];
    sv[v*64+k] = acc;
  } else {
    int r = tid-256, v = r>>7, k = r&127;
    const float* av = v ? a2d : a2s;
    float acc = 0.f;
    for(int o=0;o<64;++o) acc += av[o]*W2[o*128+k];
    sv[256 + v*128 + k] = acc;
  }
  __syncthreads();
  for(int i=tid; i<54*256; i+=512){
    int f = i>>8, r = i&255, lane = r>>3, d = r&7;
    int hi = lane>>4, l = lane&15;
    float b0, b1f;
    if (f < 18){
      int t=f>>1, kf=f&1, col=t*16+l, k0=kf*32+hi*8+2*d+(d>>2)*8;
      auto g1=[&](int k)->float{
        if (col < 128) return W1[col*64+k];
        if (col < 132) return sv[(col-128)*64+k];
        return 0.f; };
      b0=g1(k0); b1f=g1(k0+1);
    } else if (f < 38){
      int q=f-18, t=q>>2, kf=q&3, col=t*16+l, k0=kf*32+hi*8+2*d+(d>>2)*8;
      auto g2=[&](int k)->float{
        if (col < 64) return W2[col*128+k];
        if (col == 64) return sv[256+k];
        if (col == 65) return sv[384+k];
        return 0.f; };
      b0=g2(k0); b1f=g2(k0+1);
    } else {
      int q=f-38, t=q>>1, kf=q&1, col=t*16+l, k0=kf*32+hi*8+2*d+(d>>2)*8;
      b0 = Wp[col*64+k0]; b1f = Wp[col*64+k0+1];
    }
    ws[i] = pkbf_rne(b0, b1f);
  }
}

// ---------------- main kernel: one wave = SUBT graph-tiles of 4 graphs (M=16 rows each) ----------------
__global__ __launch_bounds__(64) void gat_main_kernel(
    const float* __restrict__ nodes,
    const float* __restrict__ b1, const float* __restrict__ b2, const float* __restrict__ bp,
    const uint32* __restrict__ wfrag,
    float* __restrict__ fused_out, float* __restrict__ x_out)
{
  __shared__ unsigned char lds[2*SUBT*WAVE_LDS];
  const int tid  = threadIdx.x;
  const int lane = tid & 31, wv = tid >> 5;         // 2 waves / block
  const int l16  = lane & 15, hfl = lane >> 4;
  const int wgid = blockIdx.x*2 + wv;               // global wave id
  unsigned char* wbase = lds + wv*(SUBT*WAVE_LDS);  // region s: wbase + s*WAVE_LDS
  const uint4* wsf = (const uint4*)wfrag;

  if (lane < 8) *(uint32*)(wbase + ZP_OFF + lane*4) = 0u;  // shared zero pad (s=0 region)
  const unsigned char* zp = wbase + ZP_OFF;

  float b1v[8], b2v[4], bpv[8];
  #pragma unroll
  for(int t=0;t<8;++t) b1v[t] = b1[t*16+l16];
  #pragma unroll
  for(int t=0;t<4;++t) b2v[t] = b2[t*16+l16];
  #pragma unroll
  for(int t=0;t<8;++t) bpv[t] = bp[t*16+l16];

  // A1 frags per sub-tile
  u32x8 A1[SUBT][2];
  #pragma unroll
  for(int s=0;s<SUBT;++s){
    const float* xrow = nodes + ((wgid*SUBT + s)*16 + l16)*64;
    #pragma unroll
    for(int kf=0;kf<2;++kf){
      int w0 = 32*kf + 8*hfl;
      float4 f0 = *(const float4*)(xrow + w0);
      float4 f1 = *(const float4*)(xrow + w0 + 4);
      float4 f2 = *(const float4*)(xrow + w0 + 16);
      float4 f3 = *(const float4*)(xrow + w0 + 20);
      u32x8 a;
      a[0]=pkbf(f0.x,f0.y); a[1]=pkbf(f0.z,f0.w);
      a[2]=pkbf(f1.x,f1.y); a[3]=pkbf(f1.z,f1.w);
      a[4]=pkbf(f2.x,f2.y); a[5]=pkbf(f2.z,f2.w);
      a[6]=pkbf(f3.x,f3.y); a[7]=pkbf(f3.z,f3.w);
      A1[s][kf]=a;
    }
  }
  fence_lds();

  // ---- Layer 1: shared B-frag load, WMMA for both sub-tiles ----
  #pragma unroll
  for(int t=0;t<9;++t){
    const uint4* p0 = wsf + (t*2+0)*64 + lane*2;
    const uint4* p1 = wsf + (t*2+1)*64 + lane*2;
    u32x8 Bf0 = mk8(p0[0], p0[1]);
    u32x8 Bf1 = mk8(p1[0], p1[1]);
    #pragma unroll
    for(int s=0;s<SUBT;++s){
      unsigned char* w = wbase + s*WAVE_LDS;
      v8f c = {0.f,0.f,0.f,0.f,0.f,0.f,0.f,0.f};
      c = wmma_bf16(A1[s][0], Bf0, c);
      c = wmma_bf16(A1[s][1], Bf1, c);
      if (t < 8){
        uint4 pk4;
        pk4.x = pkbf(c[0],c[1]); pk4.y = pkbf(c[2],c[3]);
        pk4.z = pkbf(c[4],c[5]); pk4.w = pkbf(c[6],c[7]);
        *(uint4*)(w + CM_OFF + (t*16+l16)*CM_STRIDE + hfl*16) = pk4;
      } else {
        unsigned char* sb = w + SC_OFF + l16*SC_STRIDE + hfl*32;
        #pragma unroll
        for(int r=0;r<8;++r) *(float*)(sb + r*4) = c[r];
      }
    }
  }
  fence_lds();

  // ---- attention-1 softmax -> alpha A-frags (per sub-tile, both heads) ----
  const int g = l16 >> 2;
  u32x8 AF[SUBT][2];
  #pragma unroll
  for(int s=0;s<SUBT;++s){
    unsigned char* w = wbase + s*WAVE_LDS;
    #pragma unroll
    for(int h=0;h<2;++h){
      float sd = *(const float*)(w + SC_OFF + (2*h+1)*SC_STRIDE + l16*4);
      float e[4];
      #pragma unroll
      for(int j=0;j<4;++j){
        float ss = *(const float*)(w + SC_OFF + (2*h)*SC_STRIDE + ((l16&12)+j)*4);
        float v = sd + ss;
        e[j] = v > 0.f ? v : 0.2f*v;       // leaky_relu 0.2
      }
      float m = fmaxf(fmaxf(e[0],e[1]), fmaxf(e[2],e[3]));
      float p0=__expf(e[0]-m), p1=__expf(e[1]-m), p2=__expf(e[2]-m), p3=__expf(e[3]-m);
      float inv = 1.f/(p0+p1+p2+p3);
      uint32 q01 = pkbf(p0*inv, p1*inv), q23 = pkbf(p2*inv, p3*inv);
      u32x8 af = {0,0,0,0,0,0,0,0};
      if (hfl == 0){
        af[2*g]=q01; af[2*g+1]=q23;
      }
      AF[s][h]=af;
    }
  }

  // ---- aggregation-1 (alpha @ xw, K=16 zero-padded WMMA) + bias + elu -> x2 rowmaj ----
  #pragma unroll
  for(int s=0;s<SUBT;++s){
    unsigned char* w = wbase + s*WAVE_LDS;
    #pragma unroll
    for(int h=0;h<2;++h){
      #pragma unroll
      for(int tt=0;tt<4;++tt){
        int tc = h*64 + tt*16;
        const unsigned char* cb = (hfl==0) ? (w + CM_OFF + (tc + l16)*CM_STRIDE) : zp;
        const uint4* bq = (const uint4*)cb;
        u32x8 Bf = mk8(bq[0], bq[1]);
        v8f c = {0.f,0.f,0.f,0.f,0.f,0.f,0.f,0.f};
        c = wmma_bf16(AF[s][h], Bf, c);
        float bb = b1v[h*4+tt];
        unsigned char* rb = w + RM_OFF + (8*hfl)*RM_PITCH + (tc + l16)*2;
        #pragma unroll
        for(int r=0;r<8;++r){
          float f = c[r] + bb;
          f = f > 0.f ? f : (__expf(f) - 1.f);   // elu
          *(unsigned short*)(rb + r*RM_PITCH) = bf16b(f);
        }
      }
    }
  }
  fence_lds();

  // ---- A2 frags from rowmaj x2 ----
  u32x8 A2[SUBT][4];
  #pragma unroll
  for(int s=0;s<SUBT;++s){
    unsigned char* w = wbase + s*WAVE_LDS;
    #pragma unroll
    for(int kf=0;kf<4;++kf){
      const unsigned char* ab = w + RM_OFF + l16*RM_PITCH + (32*kf + 8*hfl)*2;
      A2[s][kf] = mk8(*(const uint4*)ab, *(const uint4*)(ab+32));
    }
  }

  // ---- Layer 2: shared 4 B-frags per tile, both sub-tiles ----
  #pragma unroll
  for(int t=0;t<5;++t){
    u32x8 Bf[4];
    #pragma unroll
    for(int kf=0;kf<4;++kf){
      const uint4* p = wsf + (18 + t*4 + kf)*64 + lane*2;
      Bf[kf] = mk8(p[0], p[1]);
    }
    #pragma unroll
    for(int s=0;s<SUBT;++s){
      unsigned char* w = wbase + s*WAVE_LDS;
      v8f c = {0.f,0.f,0.f,0.f,0.f,0.f,0.f,0.f};
      #pragma unroll
      for(int kf=0;kf<4;++kf) c = wmma_bf16(A2[s][kf], Bf[kf], c);
      if (t < 4){
        uint4 pk4;
        pk4.x = pkbf(c[0],c[1]); pk4.y = pkbf(c[2],c[3]);
        pk4.z = pkbf(c[4],c[5]); pk4.w = pkbf(c[6],c[7]);
        *(uint4*)(w + CM_OFF + (t*16+l16)*CM_STRIDE + hfl*16) = pk4;
      } else {
        unsigned char* sb = w + SC_OFF + l16*SC_STRIDE + hfl*32;
        #pragma unroll
        for(int r=0;r<8;++r) *(float*)(sb + r*4) = c[r];
      }
    }
  }
  fence_lds();

  // ---- attention-2 softmax (1 head) ----
  u32x8 AF2[SUBT];
  #pragma unroll
  for(int s=0;s<SUBT;++s){
    unsigned char* w = wbase + s*WAVE_LDS;
    float sd = *(const float*)(w + SC_OFF + 1*SC_STRIDE + l16*4);
    float e[4];
    #pragma unroll
    for(int j=0;j<4;++j){
      float ss = *(const float*)(w + SC_OFF + 0*SC_STRIDE + ((l16&12)+j)*4);
      float v = sd + ss;
      e[j] = v > 0.f ? v : 0.2f*v;
    }
    float m = fmaxf(fmaxf(e[0],e[1]), fmaxf(e[2],e[3]));
    float p0=__expf(e[0]-m), p1=__expf(e[1]-m), p2=__expf(e[2]-m), p3=__expf(e[3]-m);
    float inv = 1.f/(p0+p1+p2+p3);
    uint32 q01 = pkbf(p0*inv, p1*inv), q23 = pkbf(p2*inv, p3*inv);
    u32x8 af = {0,0,0,0,0,0,0,0};
    if (hfl == 0){
      af[2*g]=q01; af[2*g+1]=q23;
    }
    AF2[s] = af;
  }

  // ---- aggregation-2 + bias + elu -> x3: global x output + rowmaj for projection ----
  #pragma unroll
  for(int s=0;s<SUBT;++s){
    unsigned char* w = wbase + s*WAVE_LDS;
    float* xo = x_out + ((wgid*SUBT + s)*16 + 8*hfl)*64;
    #pragma unroll
    for(int tt=0;tt<4;++tt){
      const unsigned char* cb = (hfl==0) ? (w + CM_OFF + (tt*16 + l16)*CM_STRIDE) : zp;
      const uint4* bq = (const uint4*)cb;
      u32x8 Bf = mk8(bq[0], bq[1]);
      v8f c = {0.f,0.f,0.f,0.f,0.f,0.f,0.f,0.f};
      c = wmma_bf16(AF2[s], Bf, c);
      float bb = b2v[tt];
      unsigned char* rb = w + RM_OFF + (8*hfl)*RM_PITCH + (tt*16 + l16)*2;
      #pragma unroll
      for(int r=0;r<8;++r){
        float f = c[r] + bb;
        f = f > 0.f ? f : (__expf(f) - 1.f);
        xo[r*64 + tt*16 + l16] = f;
        *(unsigned short*)(rb + r*RM_PITCH) = bf16b(f);
      }
    }
  }
  fence_lds();

  // ---- A3 frags from rowmaj x3 ----
  u32x8 A3[SUBT][2];
  #pragma unroll
  for(int s=0;s<SUBT;++s){
    unsigned char* w = wbase + s*WAVE_LDS;
    #pragma unroll
    for(int kf=0;kf<2;++kf){
      const unsigned char* ab = w + RM_OFF + l16*RM_PITCH + (32*kf + 8*hfl)*2;
      A3[s][kf] = mk8(*(const uint4*)ab, *(const uint4*)(ab+32));
    }
  }

  // ---- projection + per-graph mean + bias ----
  #pragma unroll
  for(int t=0;t<8;++t){
    const uint4* p0 = wsf + (38 + t*2 + 0)*64 + lane*2;
    const uint4* p1 = wsf + (38 + t*2 + 1)*64 + lane*2;
    u32x8 Bf0 = mk8(p0[0], p0[1]);
    u32x8 Bf1 = mk8(p1[0], p1[1]);
    #pragma unroll
    for(int s=0;s<SUBT;++s){
      v8f c = {0.f,0.f,0.f,0.f,0.f,0.f,0.f,0.f};
      c = wmma_bf16(A3[s][0], Bf0, c);
      c = wmma_bf16(A3[s][1], Bf1, c);
      float lo = (c[0]+c[1]+c[2]+c[3])*0.25f + bpv[t];
      float hi = (c[4]+c[5]+c[6]+c[7])*0.25f + bpv[t];
      float* fo = fused_out + ((wgid*SUBT + s)*4 + hfl*2)*128;
      fo[t*16 + l16]       = lo;
      fo[128 + t*16 + l16] = hi;
    }
  }
}

extern "C" void kernel_launch(void* const* d_in, const int* in_sizes, int n_in,
                              void* d_out, int out_size, void* d_ws, size_t ws_size,
                              hipStream_t stream)
{
  (void)in_sizes; (void)n_in; (void)out_size; (void)ws_size;
  const float* nodes = (const float*)d_in[0];
  const float* W1    = (const float*)d_in[1];
  const float* a1s   = (const float*)d_in[2];
  const float* a1d   = (const float*)d_in[3];
  const float* b1    = (const float*)d_in[4];
  const float* W2    = (const float*)d_in[5];
  const float* a2s   = (const float*)d_in[6];
  const float* a2d   = (const float*)d_in[7];
  const float* b2    = (const float*)d_in[8];
  const float* Wp    = (const float*)d_in[9];
  const float* bp    = (const float*)d_in[10];
  float* fused = (float*)d_out;
  float* xout  = fused + (size_t)BGRAPH*128;
  uint32* ws   = (uint32*)d_ws;

  hipLaunchKernelGGL(gat_prep_kernel, dim3(1), dim3(512), 0, stream,
                     W1, a1s, a1d, W2, a2s, a2d, Wp, ws);
  // waves = B / (4 graphs * SUBT) = 32768 ; 2 waves per block
  hipLaunchKernelGGL(gat_main_kernel, dim3(BGRAPH/(4*SUBT*2)), dim3(64), 0, stream,
                     nodes, b1, b2, bp, ws, fused, xout);
}
